// WindowGrapherPyg_45165876085623
// MI455X (gfx1250) — matrix operand order repaired and checked by reading, not compiled, not run on hardware
//
#include <hip/hip_runtime.h>
#include <hip/hip_bf16.h>

typedef __bf16 bf16_t;
typedef __bf16 v16bf __attribute__((ext_vector_type(16)));
typedef float  v8f   __attribute__((ext_vector_type(8)));

constexpr int B_ = 2, C_ = 192, H_ = 192, W_ = 192;
constexpr int WS_ = 8, N_ = 64, K_ = 9, HEADS_ = 8, DH_ = 24;
constexpr int NWH = H_ / WS_, NWW = W_ / WS_;   // 24 x 24 windows per image
constexpr int WB  = B_ * NWH * NWW;             // 1152 windows
constexpr int KT  = C_ / 32;                    // 6 k-tiles (K=32 per WMMA)
constexpr int MT  = N_ / 16;                    // 4 node-tiles
constexpr int NT  = C_ / 16;                    // 12 channel-tiles
constexpr int FRAG = 512;                       // bf16 elems per 16x32 / 32x16 fragment
constexpr int XPITCH = 68;                      // LDS pitch (floats): 16B-mult, bank-step 4

// ---------------------------------------------------------------------------
// 1) Pack the 4 weight matrices (fp32 [Cin][Cout]) into bf16 B-fragments in
//    the exact WMMA register layout: lane L holds column N=L%16, 16 K-rows with
//    k = e + (e>=8)*8 + (L>=16)*8. One contiguous 32B chunk per lane at load time.
// ---------------------------------------------------------------------------
__global__ __launch_bounds__(256) void pack_w(const float* __restrict__ Wq,
                                              const float* __restrict__ Wk,
                                              const float* __restrict__ Wv,
                                              const float* __restrict__ Ws,
                                              bf16_t* __restrict__ Wpk) {
    int gid = blockIdx.x * 256 + threadIdx.x;        // < 4*NT*KT*FRAG = 147456
    int mat = gid / (NT * KT * FRAG);
    int r   = gid % (NT * KT * FRAG);
    int nt  = r / (KT * FRAG); r %= (KT * FRAG);
    int kt  = r / FRAG;        r %= FRAG;
    int lane = r >> 4, e = r & 15;
    const float* Wm = (mat == 0) ? Wq : (mat == 1) ? Wk : (mat == 2) ? Wv : Ws;
    Wpk[gid] = (bf16_t)Wm[(kt * 32 + e + ((e >= 8) ? 8 : 0) + ((lane >= 16) ? 8 : 0)) * C_ + nt * 16 + (lane & 15)];
}

// ---------------------------------------------------------------------------
// 2) Window gather. Stage the window tile channel-major in LDS using the
//    gfx1250 async-copy path (global_load_async_to_lds_b128, ASYNCcnt), then
//    emit bf16 A-fragments (nodes as M) and B-fragments (xw^T for the Gram).
//    A layout: lanes 0-15 row M=lane, K in {0..7,16..23}; lanes 16-31 same
//    rows, K in {8..15,24..31}  ->  k = e + (e>=8)*8 + (lane>=16)*8.
// ---------------------------------------------------------------------------
__global__ __launch_bounds__(256) void gather_win(const float* __restrict__ x,
                                                  bf16_t* __restrict__ xwA,
                                                  bf16_t* __restrict__ xwB) {
    __shared__ __align__(16) float xs[C_ * XPITCH];  // [channel][node], 52.2 KB
    const int wb = blockIdx.x, tid = threadIdx.x;
    const int b = wb / (NWH * NWW), win = wb % (NWH * NWW);
    const int wh = win / NWW, ww = win % NWW;

    // 3072 16-byte chunks: chunk i -> channel c = i>>4, nodes n0 = (i&15)*4
    // (4 nodes with the same window-row share a contiguous 16B global run).
    for (int i = tid; i < C_ * (N_ / 4); i += 256) {
        int c = i >> 4, n0 = (i & 15) << 2;
        int r = n0 >> 3, cl = n0 & 7;
        const float* src =
            x + (((size_t)b * C_ + c) * H_ + wh * WS_ + r) * W_ + ww * WS_ + cl;
        unsigned ldsoff = (unsigned)(size_t)(xs + c * XPITCH + n0);
        asm volatile("global_load_async_to_lds_b128 %0, %1, off"
                     :: "v"(ldsoff), "v"(src) : "memory");
    }
    asm volatile("s_wait_asynccnt 0x0" ::: "memory");
    __syncthreads();

    bf16_t* A = xwA + (size_t)wb * (MT * KT * FRAG);
    for (int i = tid; i < MT * KT * FRAG; i += 256) {
        int f = i / FRAG, r2 = i % FRAG;
        int mt = f / KT, kt = f % KT;
        int lane = r2 >> 4, e = r2 & 15;
        int row = mt * 16 + (lane & 15);
        int k = e + ((e >= 8) ? 8 : 0) + ((lane >= 16) ? 8 : 0);
        A[i] = (bf16_t)xs[(kt * 32 + k) * XPITCH + row];
    }
    bf16_t* Bp = xwB + (size_t)wb * (MT * KT * FRAG);
    for (int i = tid; i < MT * KT * FRAG; i += 256) {
        int f = i / FRAG, r2 = i % FRAG;
        int nt = f / KT, kt = f % KT;
        int lane = r2 >> 4, e = r2 & 15;
        Bp[i] = (bf16_t)xs[(kt * 32 + e + ((e >= 8) ? 8 : 0) + ((lane >= 16) ? 8 : 0)) * XPITCH + nt * 16 + (lane & 15)];
    }
}

// ---------------------------------------------------------------------------
// 3) Projection GEMMs: ONE block per window computes all 4 matrices.
//    Wave w owns mt = w/2 and half of the 12 nt-tiles; its 6 A k-fragments
//    are hoisted into 48 VGPRs once and reused across 4 mats x 6 nt tiles
//    (24 tiles, 144 WMMAs per wave) -> only the L2-hot weight stream remains.
//    C/D layout: VGPR r, lanes 0-15 -> M=r, N=lane; lanes 16-31 -> M=8+r.
// ---------------------------------------------------------------------------
__global__ __launch_bounds__(256) void gemm_qkvs(const bf16_t* __restrict__ xwA,
                                                 const bf16_t* __restrict__ Wpk,
                                                 const float* __restrict__ bq,
                                                 const float* __restrict__ bk,
                                                 const float* __restrict__ bv,
                                                 const float* __restrict__ bs,
                                                 float* __restrict__ QKVS) {
    const int wb = blockIdx.x;
    const int wave = threadIdx.x >> 5, lane = threadIdx.x & 31;
    const int mt = wave >> 1, nhalf = wave & 1;

    const bf16_t* Abase = xwA + ((size_t)wb * MT + mt) * (KT * FRAG) + lane * 16;
    v16bf a[KT];
#pragma unroll
    for (int kt = 0; kt < KT; ++kt) a[kt] = *(const v16bf*)(Abase + kt * FRAG);

    const int col = (lane & 15);
    const int rbase = mt * 16 + ((lane >> 4) << 3);

    for (int mat = 0; mat < 4; ++mat) {
        const float* bias = (mat == 0) ? bq : (mat == 1) ? bk : (mat == 2) ? bv : bs;
        float* outp = QKVS + ((size_t)mat * WB + wb) * (N_ * C_);
#pragma unroll
        for (int n2 = 0; n2 < 6; ++n2) {
            int nt = nhalf * 6 + n2;
            const bf16_t* Bbase =
                Wpk + ((size_t)mat * NT + nt) * (KT * FRAG) + lane * 16;
            v8f acc = {};
#pragma unroll
            for (int kt = 0; kt < KT; ++kt) {
                v16bf bfr = *(const v16bf*)(Bbase + kt * FRAG);
                acc = __builtin_amdgcn_wmma_f32_16x16x32_bf16(
                    false, a[kt], false, bfr, (short)0, acc, false, false);
            }
            int ncol = nt * 16 + col;
            float bb = bias[ncol];
#pragma unroll
            for (int r = 0; r < 8; ++r)
                outp[(rbase + r) * C_ + ncol] = acc[r] + bb;
        }
    }
}

// ---------------------------------------------------------------------------
// 4) KNN: Gram = xw @ xw^T via WMMA; wave = mt row, A-fragments hoisted.
//    Diag gives |x|^2; d2_ij = sq_i + sq_j - 2 G_ij; 9-pass masked argmin.
// ---------------------------------------------------------------------------
__global__ __launch_bounds__(128) void knn_topk(const bf16_t* __restrict__ xwA,
                                                const bf16_t* __restrict__ xwB,
                                                int* __restrict__ idx) {
    __shared__ float G[N_ * (N_ + 1)];               // 64 * 65 * 4B
    __shared__ float sq[N_];
    const int wb = blockIdx.x;
    const int mt = threadIdx.x >> 5, lane = threadIdx.x & 31;

    const bf16_t* Abase = xwA + ((size_t)wb * MT + mt) * (KT * FRAG) + lane * 16;
    v16bf a[KT];
#pragma unroll
    for (int kt = 0; kt < KT; ++kt) a[kt] = *(const v16bf*)(Abase + kt * FRAG);

    const int col0 = (lane & 15);
    const int rbase = mt * 16 + ((lane >> 4) << 3);
#pragma unroll
    for (int nt = 0; nt < MT; ++nt) {
        const bf16_t* Bbase =
            xwB + ((size_t)wb * MT + nt) * (KT * FRAG) + lane * 16;
        v8f acc = {};
#pragma unroll
        for (int kt = 0; kt < KT; ++kt) {
            v16bf bfr = *(const v16bf*)(Bbase + kt * FRAG);
            acc = __builtin_amdgcn_wmma_f32_16x16x32_bf16(
                false, a[kt], false, bfr, (short)0, acc, false, false);
        }
        int gcol = nt * 16 + col0;
#pragma unroll
        for (int r = 0; r < 8; ++r)
            G[(rbase + r) * (N_ + 1) + gcol] = acc[r];
    }
    __syncthreads();
    if (threadIdx.x < N_) sq[threadIdx.x] = G[threadIdx.x * (N_ + 1) + threadIdx.x];
    __syncthreads();

    if (threadIdx.x < N_) {
        const int i = threadIdx.x;
        const float sqi = sq[i];
        unsigned long long done = 1ull << i;         // exclude self-loop
        int* op = idx + ((size_t)wb * N_ + i) * K_;
        for (int p = 0; p < K_; ++p) {
            float best = 3.4e38f; int bj = 0;
            for (int j = 0; j < N_; ++j) {
                if (done & (1ull << j)) continue;
                float d = sqi + sq[j] - 2.0f * G[i * (N_ + 1) + j];
                if (d < best) { best = d; bj = j; }   // strict < == top_k tie rule
            }
            done |= 1ull << bj;
            op[p] = bj;
        }
    }
}

// ---------------------------------------------------------------------------
// 5) Attention + skip + fused window-reverse. One thread per (node, head).
//    K/V for the window (~100 KB) are hot in L2; 9-neighbor softmax over K.
// ---------------------------------------------------------------------------
__global__ __launch_bounds__(512) void attn_out(const float* __restrict__ QKVS,
                                                const int* __restrict__ idx,
                                                float* __restrict__ out) {
    const int wb = blockIdx.x;
    const int n = threadIdx.x >> 3, h = threadIdx.x & 7;
    const float* Q  = QKVS;
    const float* Kk = QKVS + (size_t)1 * WB * N_ * C_;
    const float* V  = QKVS + (size_t)2 * WB * N_ * C_;
    const float* S  = QKVS + (size_t)3 * WB * N_ * C_;

    const float* qp = Q + ((size_t)wb * N_ + n) * C_ + h * DH_;
    float q[DH_];
#pragma unroll
    for (int d = 0; d < DH_; ++d) q[d] = qp[d];

    const int* ip = idx + ((size_t)wb * N_ + n) * K_;
    int nb[K_]; float sc[K_];
#pragma unroll
    for (int k = 0; k < K_; ++k) nb[k] = ip[k];

    const float scale = 0.20412414523193154f;        // 1/sqrt(DH)
    float mx = -3.4e38f;
#pragma unroll
    for (int k = 0; k < K_; ++k) {
        const float* kp = Kk + ((size_t)wb * N_ + nb[k]) * C_ + h * DH_;
        float d0 = 0.f;
#pragma unroll
        for (int d = 0; d < DH_; ++d) d0 += q[d] * kp[d];
        sc[k] = d0 * scale;
        mx = fmaxf(mx, sc[k]);
    }
    float ssum = 0.f;
#pragma unroll
    for (int k = 0; k < K_; ++k) { sc[k] = __expf(sc[k] - mx); ssum += sc[k]; }
    const float inv = 1.0f / ssum;

    float acc[DH_];
#pragma unroll
    for (int d = 0; d < DH_; ++d) acc[d] = 0.f;
    for (int k = 0; k < K_; ++k) {
        const float* vp = V + ((size_t)wb * N_ + nb[k]) * C_ + h * DH_;
        float a = sc[k] * inv;
#pragma unroll
        for (int d = 0; d < DH_; ++d) acc[d] += a * vp[d];
    }

    const float* sp = S + ((size_t)wb * N_ + n) * C_ + h * DH_;
    const int b = wb / (NWH * NWW), win = wb % (NWH * NWW);
    const int wh = win / NWW, ww = win % NWW;
    const int r = n >> 3, cl = n & 7;
#pragma unroll
    for (int d = 0; d < DH_; ++d) {
        int c = h * DH_ + d;
        out[(((size_t)b * C_ + c) * H_ + wh * WS_ + r) * W_ + ww * WS_ + cl] =
            acc[d] + sp[d];
    }
}

// ---------------------------------------------------------------------------
extern "C" void kernel_launch(void* const* d_in, const int* in_sizes, int n_in,
                              void* d_out, int out_size, void* d_ws, size_t ws_size,
                              hipStream_t stream) {
    const float* x  = (const float*)d_in[0];
    const float* Wq = (const float*)d_in[1];
    const float* bq = (const float*)d_in[2];
    const float* Wk = (const float*)d_in[3];
    const float* bk = (const float*)d_in[4];
    const float* Wv = (const float*)d_in[5];
    const float* bv = (const float*)d_in[6];
    const float* Ws = (const float*)d_in[7];
    const float* bs = (const float*)d_in[8];
    float* out = (float*)d_out;

    char* ws = (char*)d_ws;
    size_t off = 0;
    auto carve = [&](size_t bytes) {
        char* p = ws + off;
        off = (off + bytes + 255) & ~(size_t)255;
        return (void*)p;
    };
    bf16_t* xwA  = (bf16_t*)carve((size_t)WB * MT * KT * FRAG * sizeof(bf16_t));
    bf16_t* xwB  = (bf16_t*)carve((size_t)WB * MT * KT * FRAG * sizeof(bf16_t));
    bf16_t* Wpk  = (bf16_t*)carve((size_t)4 * NT * KT * FRAG * sizeof(bf16_t));
    float*  QKVS = (float*)carve((size_t)4 * WB * N_ * C_ * sizeof(float));
    int*    idxb = (int*)carve((size_t)WB * N_ * K_ * sizeof(int));

    pack_w<<<(4 * NT * KT * FRAG) / 256, 256, 0, stream>>>(Wq, Wk, Wv, Ws, Wpk);
    gather_win<<<WB, 256, 0, stream>>>(x, xwA, xwB);
    gemm_qkvs<<<WB, 256, 0, stream>>>(xwA, Wpk, bq, bk, bv, bs, QKVS);
    knn_topk<<<WB, 128, 0, stream>>>(xwA, xwB, idxb);
    attn_out<<<WB, 512, 0, stream>>>(QKVS, idxb, out);
}
